// BotAwareGAT_87986700026589
// MI455X (gfx1250) — compile-verified, matrix operand order checked by reading, not run on hardware
//
#include <hip/hip_runtime.h>
#include <math.h>

#define TPB 256

typedef __attribute__((ext_vector_type(16))) __bf16 v16bf;
typedef __attribute__((ext_vector_type(8)))  float  v8f;

union U16x8 { unsigned int u[8]; v16bf v; };

__device__ __forceinline__ unsigned short f32_to_bf16_rne(float f) {
  unsigned int u = __float_as_uint(f);
  u += 0x7fffu + ((u >> 16) & 1u);
  return (unsigned short)(u >> 16);
}

__device__ __forceinline__ void atomicMaxF32(float* addr, float val) {
  unsigned int* ua = (unsigned int*)addr;
  unsigned int old = *ua;
  while (true) {
    float f = __uint_as_float(old);
    if (f >= val) break;
    unsigned int assumed = old;
    old = atomicCAS(ua, assumed, __float_as_uint(val));
    if (old == assumed) break;
  }
}

// ---------------- conversion kernels ----------------
__global__ void k_cvt_bf16(const float* __restrict__ src, unsigned short* __restrict__ dst, size_t n) {
  size_t i = (size_t)blockIdx.x * blockDim.x + threadIdx.x;
  if (i < n) dst[i] = f32_to_bf16_rne(src[i]);
}

// W [K, HC] row-major f32 -> Wt [HC, K] row-major bf16 (i.e. B in column-major)
__global__ void k_cvt_w_t(const float* __restrict__ W, unsigned short* __restrict__ Wt, int K, int HC) {
  int i = blockIdx.x * blockDim.x + threadIdx.x;
  if (i >= K * HC) return;
  int k = i / HC, c = i % HC;
  Wt[(size_t)c * K + k] = f32_to_bf16_rne(W[i]);
}

__global__ void k_fill(float* __restrict__ p, float v, size_t n) {
  size_t i = (size_t)blockIdx.x * blockDim.x + threadIdx.x;
  if (i < n) p[i] = v;
}

// ---------------- WMMA bf16 GEMM: C[M,N] = A[M,K] * B[K,N] ----------------
// A: [M,K] bf16 row-major.  Bt: [N,K] bf16 row-major (== B column-major).
// One wave computes one 16x16 tile of C with v_wmma_f32_16x16x32_bf16.
__global__ __launch_bounds__(TPB) void gemm_bf16_wmma(
    const unsigned short* __restrict__ A, const unsigned short* __restrict__ Bt,
    float* __restrict__ C, int M, int Nn, int K) {
  int wid  = threadIdx.x >> 5;
  int lane = threadIdx.x & 31;
  int l15  = lane & 15;
  int hi   = lane >> 4;           // wave32: lanes 0-15 / 16-31 halves
  int tilesN = Nn >> 4;
  int tilesM = (M + 15) >> 4;
  int tile = blockIdx.x * 8 + wid;
  int tm = tile / tilesN;
  int tn = tile - tm * tilesN;
  if (tm >= tilesM) return;       // wave-uniform exit: EXEC stays all-ones for WMMA

  int rowA = tm * 16 + l15; if (rowA >= M) rowA = M - 1;   // clamp (no EXEC divergence)
  const unsigned short* arow = A  + (size_t)rowA * K;
  const unsigned short* brow = Bt + (size_t)(tn * 16 + l15) * K;

  v8f acc = {};
  for (int k0 = 0; k0 < K; k0 += 32) {
    U16x8 a, b;
#pragma unroll
    for (int j = 0; j < 8; ++j) {
      // ISA 7.12.2 16-bit A/B operand layout: VGPR j holds K pair starting at:
      int kp = k0 + 2 * j + (hi << 3) + ((j >= 4) ? 8 : 0);
      a.u[j] = *(const unsigned int*)(arow + kp);
      b.u[j] = *(const unsigned int*)(brow + kp);
    }
    __builtin_prefetch(arow + k0 + 32, 0, 3);
    __builtin_prefetch(brow + k0 + 32, 0, 3);
    acc = __builtin_amdgcn_wmma_f32_16x16x32_bf16(
        false, a.v, false, b.v, (short)0, acc, false, false);
  }
  // C/D layout: VGPR r -> M = tm*16 + 8*hi + r, N = tn*16 + l15
  int rbase = tm * 16 + hi * 8;
  float* ccol = C + (size_t)rbase * Nn + tn * 16 + l15;
#pragma unroll
  for (int r = 0; r < 8; ++r)
    if (rbase + r < M) ccol[(size_t)r * Nn] = acc[r];
}

// ---------------- attention kernels ----------------
// per (node, head): s_src/s_dst dot products
__global__ void k_scores(const float* __restrict__ h, const float* __restrict__ a_src,
                         const float* __restrict__ a_dst, float* __restrict__ ssrc,
                         float* __restrict__ sdst, int Nn, int C) {
  int i = blockIdx.x * blockDim.x + threadIdx.x;
  if (i >= Nn * 8) return;
  int n = i >> 3, hd = i & 7;
  const float* hp = h + (size_t)n * 8 * C + (size_t)hd * C;
  const float* as = a_src + hd * C;
  const float* ad = a_dst + hd * C;
  float s0 = 0.f, s1 = 0.f;
  for (int c = 0; c < C; ++c) { float v = hp[c]; s0 += v * as[c]; s1 += v * ad[c]; }
  ssrc[i] = s0; sdst[i] = s1;
}

__global__ void k_edge_max(const long long* __restrict__ ei, const float* __restrict__ ssrc,
                           const float* __restrict__ sdst, float* __restrict__ m, int E) {
  int e = blockIdx.x * blockDim.x + threadIdx.x;
  if (e >= E) return;
  int s = (int)ei[e], d = (int)ei[(size_t)E + e];
#pragma unroll
  for (int hd = 0; hd < 8; ++hd) {
    float v = ssrc[s * 8 + hd] + sdst[d * 8 + hd];
    v = v > 0.f ? v : 0.2f * v;                    // leaky_relu(0.2)
    atomicMaxF32(&m[d * 8 + hd], v);
  }
}

__global__ void k_edge_exp(const long long* __restrict__ ei, const float* __restrict__ ssrc,
                           const float* __restrict__ sdst, const float* __restrict__ m,
                           float* __restrict__ pbuf, float* __restrict__ z, int E) {
  int e = blockIdx.x * blockDim.x + threadIdx.x;
  if (e >= E) return;
  int s = (int)ei[e], d = (int)ei[(size_t)E + e];
#pragma unroll
  for (int hd = 0; hd < 8; ++hd) {
    float v = ssrc[s * 8 + hd] + sdst[d * 8 + hd];
    v = v > 0.f ? v : 0.2f * v;
    float p = __expf(v - m[d * 8 + hd]);
    pbuf[(size_t)e * 8 + hd] = p;
    atomicAdd(&z[d * 8 + hd], p);
  }
}

// thread per (edge, feature): agg[dst, f] += h[src, f] * alpha(e, head(f))
__global__ void k_edge_agg(const long long* __restrict__ ei, const float* __restrict__ h,
                           const float* __restrict__ pbuf, const float* __restrict__ z,
                           float* __restrict__ agg, int E, int HC, int C) {
  size_t idx = (size_t)blockIdx.x * blockDim.x + threadIdx.x;
  if (idx >= (size_t)E * HC) return;
  int e  = (int)(idx / HC);
  int f  = (int)(idx - (size_t)e * HC);
  int hd = f / C;
  int s = (int)ei[e], d = (int)ei[(size_t)E + e];
  float alpha = pbuf[(size_t)e * 8 + hd] / z[d * 8 + hd];
  atomicAdd(&agg[(size_t)d * HC + f], h[(size_t)s * HC + f] * alpha);
}

// layer-1 epilogue: hetero-mean + bias + ELU, emit bf16 for the next WMMA GEMM
__global__ void k_combine1(const float* __restrict__ agg, const float* __restrict__ b0,
                           const float* __restrict__ b1p, unsigned short* __restrict__ out_bf,
                           int Nn) {
  int i = blockIdx.x * blockDim.x + threadIdx.x;
  if (i >= Nn * 128) return;
  int f = i & 127;
  float v = 0.5f * (agg[i] + b0[f] + b1p[f]);
  v = v > 0.f ? v : expm1f(v);                     // ELU
  out_bf[i] = f32_to_bf16_rne(v);
}

// layer-2 epilogue: head-mean + hetero-mean + bias + ELU
__global__ void k_combine2(const float* __restrict__ agg, const float* __restrict__ b0,
                           const float* __restrict__ b1p, float* __restrict__ h2, int Nn) {
  int i = blockIdx.x * blockDim.x + threadIdx.x;
  if (i >= Nn * 64) return;
  int n = i >> 6, c = i & 63;
  const float* ap = agg + (size_t)n * 512 + c;
  float s = 0.f;
#pragma unroll
  for (int hd = 0; hd < 8; ++hd) s += ap[hd * 64];
  float v = 0.5f * (s * 0.125f + b0[c] + b1p[c]);
  h2[i] = v > 0.f ? v : expm1f(v);
}

// tiny MLP head: [64]->relu[32]->[2]
__global__ void k_classifier(const float* __restrict__ h2, const float* __restrict__ Wc1,
                             const float* __restrict__ bc1, const float* __restrict__ Wc2,
                             const float* __restrict__ bc2, float* __restrict__ out, int Nn) {
  int n = blockIdx.x * blockDim.x + threadIdx.x;
  if (n >= Nn) return;
  const float* hn = h2 + (size_t)n * 64;
  float hid[32];
#pragma unroll
  for (int j = 0; j < 32; ++j) hid[j] = bc1[j];
  for (int k = 0; k < 64; ++k) {
    float hv = hn[k];
    const float* wr = Wc1 + k * 32;
#pragma unroll
    for (int j = 0; j < 32; ++j) hid[j] += hv * wr[j];
  }
  float l0 = bc2[0], l1 = bc2[1];
#pragma unroll
  for (int j = 0; j < 32; ++j) {
    float r = hid[j] > 0.f ? hid[j] : 0.f;
    l0 += r * Wc2[j * 2 + 0];
    l1 += r * Wc2[j * 2 + 1];
  }
  out[(size_t)n * 2 + 0] = l0;
  out[(size_t)n * 2 + 1] = l1;
}

// ---------------- host ----------------
static inline unsigned gridFor(size_t n) { return (unsigned)((n + TPB - 1) / TPB); }

static void run_gat_layer(const unsigned short* Abf, const unsigned short* Wbt_base, size_t wstride,
                          const float* a_src_all, const float* a_dst_all,
                          const long long* const eis[2], int Nn, int E, int K, int C,
                          float* gout, float* agg, float* ssrc, float* sdst,
                          float* m, float* z, float* pbuf, hipStream_t stream) {
  const int HC = 8 * C;
  k_fill<<<gridFor((size_t)Nn * HC), TPB, 0, stream>>>(agg, 0.f, (size_t)Nn * HC);
  const int tilesM = (Nn + 15) / 16, tilesN = HC / 16;
  const unsigned gblocks = (unsigned)((tilesM * tilesN + 7) / 8);
  for (int t = 0; t < 2; ++t) {
    gemm_bf16_wmma<<<gblocks, TPB, 0, stream>>>(Abf, Wbt_base + (size_t)t * wstride, gout, Nn, HC, K);
    k_scores<<<gridFor((size_t)Nn * 8), TPB, 0, stream>>>(
        gout, a_src_all + (size_t)t * HC, a_dst_all + (size_t)t * HC, ssrc, sdst, Nn, C);
    k_fill<<<gridFor((size_t)Nn * 8), TPB, 0, stream>>>(m, -1e30f, (size_t)Nn * 8);
    k_fill<<<gridFor((size_t)Nn * 8), TPB, 0, stream>>>(z, 0.f, (size_t)Nn * 8);
    k_edge_max<<<gridFor((size_t)E), TPB, 0, stream>>>(eis[t], ssrc, sdst, m, E);
    k_edge_exp<<<gridFor((size_t)E), TPB, 0, stream>>>(eis[t], ssrc, sdst, m, pbuf, z, E);
    k_edge_agg<<<gridFor((size_t)E * HC), TPB, 0, stream>>>(eis[t], gout, pbuf, z, agg, E, HC, C);
  }
}

extern "C" void kernel_launch(void* const* d_in, const int* in_sizes, int n_in,
                              void* d_out, int out_size, void* d_ws, size_t ws_size,
                              hipStream_t stream) {
  (void)n_in; (void)out_size; (void)ws_size;
  const float*     x   = (const float*)d_in[0];
  const long long* eiA = (const long long*)d_in[1];
  const long long* eiB = (const long long*)d_in[2];
  const float* W1  = (const float*)d_in[3];
  const float* a1s = (const float*)d_in[4];
  const float* a1d = (const float*)d_in[5];
  const float* b1  = (const float*)d_in[6];
  const float* W2  = (const float*)d_in[7];
  const float* a2s = (const float*)d_in[8];
  const float* a2d = (const float*)d_in[9];
  const float* b2  = (const float*)d_in[10];
  const float* Wc1 = (const float*)d_in[11];
  const float* bc1 = (const float*)d_in[12];
  const float* Wc2 = (const float*)d_in[13];
  const float* bc2 = (const float*)d_in[14];

  const int Nn = in_sizes[0] / 256;   // 20000
  const int E  = in_sizes[1] / 2;     // 640000
  const long long* eis[2] = { eiA, eiB };

  // workspace arena (~121 MB)
  char* ws = (char*)d_ws;
  size_t off = 0;
  auto take = [&](size_t bytes) -> char* {
    char* p = ws + off;
    off = (off + bytes + 255) & ~(size_t)255;
    return p;
  };
  unsigned short* Abf  = (unsigned short*)take((size_t)Nn * 256 * 2); // x_bf16, reused as h1_bf16
  unsigned short* Wb1  = (unsigned short*)take((size_t)2 * 256 * 128 * 2);
  unsigned short* Wb2  = (unsigned short*)take((size_t)2 * 128 * 512 * 2);
  float* gout = (float*)take((size_t)Nn * 512 * 4);
  float* agg  = (float*)take((size_t)Nn * 512 * 4);
  float* ssrc = (float*)take((size_t)Nn * 8 * 4);
  float* sdst = (float*)take((size_t)Nn * 8 * 4);
  float* mbuf = (float*)take((size_t)Nn * 8 * 4);
  float* zbuf = (float*)take((size_t)Nn * 8 * 4);
  float* pbuf = (float*)take((size_t)E * 8 * 4);
  float* h2   = (float*)take((size_t)Nn * 64 * 4);

  // prep: bf16 conversions (x, and transposed weights for the WMMA B operand)
  k_cvt_bf16<<<gridFor((size_t)Nn * 256), TPB, 0, stream>>>(x, Abf, (size_t)Nn * 256);
  for (int t = 0; t < 2; ++t) {
    k_cvt_w_t<<<gridFor((size_t)256 * 128), TPB, 0, stream>>>(
        W1 + (size_t)t * 256 * 128, Wb1 + (size_t)t * 256 * 128, 256, 128);
    k_cvt_w_t<<<gridFor((size_t)128 * 512), TPB, 0, stream>>>(
        W2 + (size_t)t * 128 * 512, Wb2 + (size_t)t * 128 * 512, 128, 512);
  }

  // layer 1: C=16, HC=128, K=256 (concat) -> h1 bf16 written back into Abf
  run_gat_layer(Abf, Wb1, (size_t)256 * 128, a1s, a1d, eis, Nn, E, 256, 16,
                gout, agg, ssrc, sdst, mbuf, zbuf, pbuf, stream);
  k_combine1<<<gridFor((size_t)Nn * 128), TPB, 0, stream>>>(agg, b1, b1 + 128, Abf, Nn);

  // layer 2: C=64, HC=512, K=128 (head mean)
  run_gat_layer(Abf, Wb2, (size_t)128 * 512, a2s, a2d, eis, Nn, E, 128, 64,
                gout, agg, ssrc, sdst, mbuf, zbuf, pbuf, stream);
  k_combine2<<<gridFor((size_t)Nn * 64), TPB, 0, stream>>>(agg, b2, b2 + 64, h2, Nn);

  // classifier head
  k_classifier<<<gridFor((size_t)Nn), TPB, 0, stream>>>(h2, Wc1, bc1, Wc2, bc2, (float*)d_out, Nn);
}
